// JointPolicy_77068893160319
// MI455X (gfx1250) — matrix operand order repaired and checked by reading, not compiled, not run on hardware
//
#include <hip/hip_runtime.h>
#include <hip/hip_bf16.h>

typedef __attribute__((ext_vector_type(16))) _Float16 v16h;
typedef __attribute__((ext_vector_type(8)))  _Float16 v8h;
typedef __attribute__((ext_vector_type(8)))  float    v8f;

#define HID   64
#define SLOTS 4
#define SEQ   24
#define TSTEPS (SEQ - 1)
#define JOINTW 320              // halves per joint row: emb(64) + mem(256)
#define ROWS_PER_WG 128
#define WAVES 8

// ---- LDS / workspace layout (bytes). ws[0..WS_WEIGHT_BYTES) is copied 1:1 into LDS.
#define LDS_W1    0             // 10 chunks * 4 tiles * 1KB = 40960
#define LDS_W2    40960         // 2*4*1KB = 8192
#define LDS_WW    49152         // 8192
#define LDS_WE    57344         // 2*1*1KB = 2048 (We padded 64x16)
#define LDS_WR1   59392         // 4*4*1KB = 16384
#define LDS_WR2   75776         // 8192
#define LDS_EMB   83968         // 66*64 f16 = 8448
#define WS_WEIGHT_BYTES 92416
#define LDS_B1    92416
#define LDS_B2    92672
#define LDS_BW    92928
#define LDS_BE    93184         // 16 f32 (padded)
#define LDS_BR1   93248
#define LDS_BR2   93504
#define LDS_JOINT 93760         // 128 rows * 320 halves * 2B = 81920
#define LDS_H     175680        // 128 * 64 f16 = 16384
#define LDS_EV    192064        // 128 * 4 f32 = 2048
#define LDS_IDX   194112        // 128 * u32 = 512
#define LDS_TOTAL 194624

// ---------------- device-side weight pre-swizzle into WMMA B-fragment layout ----
// B fragment (16x16x32 f16): VGPR v, lane n holds pair B[kb+j], B[kb+j+1] (j = 2(v&3) + 16(v>>2)) at
// column (n%16 + 16*tile), kb = 32*chunk + (n>=16 ? 8 : 0).  (ISA 05_wmma §7.12.2)
__global__ void swizzleB(const float* __restrict__ W, unsigned* __restrict__ dst,
                         int Ncols, int nT) {
    int tid  = blockIdx.x * 256 + threadIdx.x;
    int v    = tid & 7;
    int lane = (tid >> 3) & 31;
    int rest = tid >> 8;
    int t    = rest % nT;
    int c    = rest / nT;
    int k0   = 32 * c + ((lane & 16) ? 8 : 0) + 2 * (v & 3) + ((v & 4) ? 16 : 0);
    int col  = (lane & 15) + 16 * t;
    float w0 = (col < Ncols) ? W[(size_t)k0 * Ncols + col]       : 0.f;
    float w1 = (col < Ncols) ? W[(size_t)(k0 + 1) * Ncols + col] : 0.f;
    union { _Float16 h[2]; unsigned u; } p;
    p.h[0] = (_Float16)w0;
    p.h[1] = (_Float16)w1;
    dst[tid] = p.u;
}

__global__ void cvtF16(const float* __restrict__ src, _Float16* __restrict__ dst, int n) {
    int i = blockIdx.x * 256 + threadIdx.x;
    if (i < n) dst[i] = (_Float16)src[i];
}

// ---------------- fragment loads from LDS ----------------
// A (16x32 f16): lane n = row n%16; halves [hOff,hOff+8) and [16+hOff,16+hOff+8),
// hOff = (lane>=16)?8:0  -> two ds_load_b128.
__device__ __forceinline__ v16h ldsLoadA(const _Float16* p, int hOff) {
    union { v16h v; uint4 u[2]; } r;
    r.u[0] = *(const uint4*)(p + hOff);
    r.u[1] = *(const uint4*)(p + 16 + hOff);
    return r.v;
}
// B: pre-swizzled blocks of 32 lanes x 32 bytes per (chunk,tile).
__device__ __forceinline__ v16h ldsLoadB(const char* wB, int ct, int lane) {
    const uint4* p = (const uint4*)(wB + ((size_t)(ct * 32 + lane)) * 32);
    union { v16h v; uint4 u[2]; } r;
    r.u[0] = p[0];
    r.u[1] = p[1];
    return r.v;
}

template <int NT>
__device__ __forceinline__ void initBias(v8f acc[NT], const float* b, int lane) {
#pragma unroll
    for (int t = 0; t < NT; t++) {
        float bv = b[t * 16 + (lane & 15)];
#pragma unroll
        for (int i = 0; i < 8; i++) acc[t][i] = bv;
    }
}

template <int CH, int NT>
__device__ __forceinline__ void gemm(v8f acc[NT], const _Float16* aRow, int hOff,
                                     const char* wB, int lane) {
#pragma unroll
    for (int c = 0; c < CH; c++) {
        v16h a = ldsLoadA(aRow + c * 32, hOff);
#pragma unroll
        for (int t = 0; t < NT; t++) {
            v16h b = ldsLoadB(wB, c * NT + t, lane);
            acc[t] = __builtin_amdgcn_wmma_f32_16x16x32_f16(
                false, a, false, b, (short)0, acc[t], false, false);
        }
    }
}

// D layout: VGPR i, lane n -> row i + (n>=16?8:0), col tile*16 + n%16.
template <int NT>
__device__ __forceinline__ void storeReluF16(const v8f acc[NT], _Float16* dst,
                                             int stride, int lane) {
    int hi8 = (lane & 16) ? 8 : 0;
    int c0  = lane & 15;
#pragma unroll
    for (int t = 0; t < NT; t++) {
#pragma unroll
        for (int i = 0; i < 8; i++) {
            float v = acc[t][i];
            v = v > 0.f ? v : 0.f;
            dst[(i + hi8) * stride + t * 16 + c0] = (_Float16)v;
        }
    }
}

// ---------------- main fused recurrent kernel ----------------
__global__ void __launch_bounds__(256, 1)
memnet_kernel(const int* __restrict__ seqs, const int* __restrict__ qtok,
              const uint4* __restrict__ ws,
              const float* __restrict__ b1, const float* __restrict__ b2,
              const float* __restrict__ bw, const float* __restrict__ be,
              const float* __restrict__ br1, const float* __restrict__ br2,
              float* __restrict__ out) {
    extern __shared__ __align__(16) char S[];

    const char*  sW1  = S + LDS_W1;
    const char*  sW2  = S + LDS_W2;
    const char*  sWW  = S + LDS_WW;
    const char*  sWE  = S + LDS_WE;
    const char*  sWR1 = S + LDS_WR1;
    const char*  sWR2 = S + LDS_WR2;
    _Float16*    sEmb = (_Float16*)(S + LDS_EMB);
    float* sB1  = (float*)(S + LDS_B1);
    float* sB2  = (float*)(S + LDS_B2);
    float* sBW  = (float*)(S + LDS_BW);
    float* sBE  = (float*)(S + LDS_BE);
    float* sBR1 = (float*)(S + LDS_BR1);
    float* sBR2 = (float*)(S + LDS_BR2);
    _Float16* sJoint = (_Float16*)(S + LDS_JOINT);
    _Float16* sH     = (_Float16*)(S + LDS_H);
    float*    sEv    = (float*)(S + LDS_EV);
    unsigned* sIdx   = (unsigned*)(S + LDS_IDX);

    // ---- stage swizzled weights + f16 embed table from workspace into LDS
    {
        uint4* d = (uint4*)S;
        for (int i = threadIdx.x; i < WS_WEIGHT_BYTES / 16; i += 256) d[i] = ws[i];
        for (int i = threadIdx.x; i < HID; i += 256) {
            sB1[i] = b1[i]; sB2[i] = b2[i]; sBW[i] = bw[i];
            sBR1[i] = br1[i]; sBR2[i] = br2[i];
        }
        if (threadIdx.x < 16) sBE[threadIdx.x] = (threadIdx.x < SLOTS) ? be[threadIdx.x] : 0.f;
        uint4 z = {0, 0, 0, 0};
        uint4* jz = (uint4*)(S + LDS_JOINT);
        for (int i = threadIdx.x; i < ROWS_PER_WG * JOINTW * 2 / 16; i += 256) jz[i] = z;
    }
    __syncthreads();  // only barrier: rows are wave-private afterwards

    const int lane  = threadIdx.x & 31;
    const int warp  = threadIdx.x >> 5;
    const int r0loc = warp * 16;
    const int r0    = blockIdx.x * ROWS_PER_WG + r0loc;
    const int rowA  = lane & 15;              // A-matrix row this lane serves
    const int hOff  = (lane & 16) ? 8 : 0;    // A half-offset (halves)
    const int hi8   = (lane & 16) ? 8 : 0;    // D row offset
    const int half  = lane >> 4;              // 0/1: which 32-half slice of a row

    _Float16* jRow = sJoint + (size_t)(r0loc + rowA) * JOINTW;
    _Float16* hRow = sH + (size_t)(r0loc + rowA) * HID;

    for (int t = 0; t < TSTEPS; t++) {
        // 1) gather this step's token embedding into joint[0:64)
        int tok = seqs[(size_t)(r0 + rowA) * SEQ + t];
        {
            const uint4* src = (const uint4*)(sEmb + tok * HID + half * 32);
            uint4* dst = (uint4*)(jRow + half * 32);
            dst[0] = src[0];
            dst[1] = src[1];
        }
        // 2) h = relu(joint @ W1 + b1)     [16x320]@[320x64]
        v8f a1[4];
        initBias<4>(a1, sB1, lane);
        gemm<10, 4>(a1, jRow, hOff, sW1, lane);
        storeReluF16<4>(a1, sH + (size_t)r0loc * HID, HID, lane);
        // 3) h = relu(h @ W2 + b2)
        v8f a2[4];
        initBias<4>(a2, sB2, lane);
        gemm<2, 4>(a2, hRow, hOff, sW2, lane);
        storeReluF16<4>(a2, sH + (size_t)r0loc * HID, HID, lane);
        // 4) write_vec = h @ Ww + bw   (kept in accumulators)
        v8f wv[4];
        initBias<4>(wv, sBW, lane);
        gemm<2, 4>(wv, hRow, hOff, sWW, lane);
        // 5) evict logits = h @ WePad + bePad  (one padded 16-col tile)
        v8f ev[1];
        initBias<1>(ev, sBE, lane);
        gemm<2, 1>(ev, hRow, hOff, sWE, lane);
        if ((lane & 15) < SLOTS) {
#pragma unroll
            for (int i = 0; i < 8; i++)
                sEv[(r0loc + i + hi8) * SLOTS + (lane & 15)] = ev[0][i];
        }
        // 6) per-row argmax (first max wins, like jnp.argmax)
        {
            const float4 lg = *(const float4*)(sEv + (r0loc + rowA) * SLOTS);
            int   idx = 0;
            float m = lg.x;
            if (lg.y > m) { m = lg.y; idx = 1; }
            if (lg.z > m) { m = lg.z; idx = 2; }
            if (lg.w > m) { m = lg.w; idx = 3; }
            sIdx[r0loc + rowA] = (unsigned)idx;
        }
        // 7) scatter write_vec row into mem[evict_idx]
#pragma unroll
        for (int i = 0; i < 8; i++) {
            int rr = r0loc + i + hi8;
            unsigned s = sIdx[rr];
            _Float16* m = sJoint + (size_t)rr * JOINTW + HID + s * HID;
#pragma unroll
            for (int tt = 0; tt < 4; tt++)
                m[tt * 16 + (lane & 15)] = (_Float16)wv[tt][i];
        }
    }

    // ---- readout: r = [q_emb | mean(mem)] ; logits = relu(r@Wr1+br1)@Wr2+br2
    {
        int qt = qtok[r0 + rowA];
        const uint4* src = (const uint4*)(sEmb + qt * HID + half * 32);
        uint4* dst = (uint4*)(jRow + half * 32);
        dst[0] = src[0];
        dst[1] = src[1];
        // mem summary written over slot-0 region -> joint row becomes [q_emb|summary]
#pragma unroll
        for (int g = 0; g < 4; g++) {
            int c = half * 32 + g * 8;
            union { uint4 u; v8h v; } s0, s1, s2, s3, r;
            s0.u = *(const uint4*)(jRow + HID + 0 * HID + c);
            s1.u = *(const uint4*)(jRow + HID + 1 * HID + c);
            s2.u = *(const uint4*)(jRow + HID + 2 * HID + c);
            s3.u = *(const uint4*)(jRow + HID + 3 * HID + c);
            v8h sum = (s0.v + s1.v) + (s2.v + s3.v);
#pragma unroll
            for (int k = 0; k < 8; k++) r.v[k] = sum[k] * (_Float16)0.25f;
            *(uint4*)(jRow + HID + c) = r.u;
        }
        v8f h1[4];
        initBias<4>(h1, sBR1, lane);
        gemm<4, 4>(h1, jRow, hOff, sWR1, lane);     // [16x128]@[128x64]
        storeReluF16<4>(h1, sH + (size_t)r0loc * HID, HID, lane);
        v8f o[4];
        initBias<4>(o, sBR2, lane);
        gemm<2, 4>(o, hRow, hOff, sWR2, lane);      // [16x64]@[64x64]
#pragma unroll
        for (int tt = 0; tt < 4; tt++)
#pragma unroll
            for (int i = 0; i < 8; i++)
                out[(size_t)(r0 + i + hi8) * HID + tt * 16 + (lane & 15)] = o[tt][i];
    }
}

extern "C" void kernel_launch(void* const* d_in, const int* in_sizes, int n_in,
                              void* d_out, int out_size, void* d_ws, size_t ws_size,
                              hipStream_t stream) {
    const int*   seqs  = (const int*)d_in[0];
    const int*   qtok  = (const int*)d_in[1];
    const float* embed = (const float*)d_in[2];
    const float* W1 = (const float*)d_in[3];
    const float* b1 = (const float*)d_in[4];
    const float* W2 = (const float*)d_in[5];
    const float* b2 = (const float*)d_in[6];
    const float* Ww = (const float*)d_in[7];
    const float* bw = (const float*)d_in[8];
    const float* We = (const float*)d_in[9];
    const float* be = (const float*)d_in[10];
    const float* Wr1 = (const float*)d_in[11];
    const float* br1 = (const float*)d_in[12];
    const float* Wr2 = (const float*)d_in[13];
    const float* br2 = (const float*)d_in[14];
    float* out = (float*)d_out;
    char*  ws  = (char*)d_ws;

    // pre-swizzle all weights into WMMA B-fragment layout (f16) + f16 embed table
    swizzleB<<<40, 256, 0, stream>>>(W1, (unsigned*)(ws + LDS_W1), 64, 4);   // 320x64
    swizzleB<<<8,  256, 0, stream>>>(W2, (unsigned*)(ws + LDS_W2), 64, 4);   // 64x64
    swizzleB<<<8,  256, 0, stream>>>(Ww, (unsigned*)(ws + LDS_WW), 64, 4);   // 64x64
    swizzleB<<<2,  256, 0, stream>>>(We, (unsigned*)(ws + LDS_WE), 4, 1);    // 64x4 -> pad 64x16
    swizzleB<<<16, 256, 0, stream>>>(Wr1, (unsigned*)(ws + LDS_WR1), 64, 4); // 128x64
    swizzleB<<<8,  256, 0, stream>>>(Wr2, (unsigned*)(ws + LDS_WR2), 64, 4); // 64x64
    cvtF16<<<(66 * HID + 255) / 256, 256, 0, stream>>>(embed, (_Float16*)(ws + LDS_EMB), 66 * HID);

    int B = in_sizes[0] / SEQ;
    hipFuncSetAttribute((const void*)memnet_kernel,
                        hipFuncAttributeMaxDynamicSharedMemorySize, LDS_TOTAL);
    memnet_kernel<<<B / ROWS_PER_WG, 256, LDS_TOTAL, stream>>>(
        seqs, qtok, (const uint4*)ws, b1, b2, bw, be, br1, br2, out);
}
